// CustomRNN_83760452207459
// MI455X (gfx1250) — compile-verified, matrix operand order checked by reading, not run on hardware
//
#include <hip/hip_runtime.h>
#include <math.h>

#define Bsz 256
#define Tsz 256
#define Dsz 1024
#define Hsz 1024

typedef __attribute__((ext_vector_type(16))) __bf16 v16bf;
typedef __attribute__((ext_vector_type(8)))  float  v8f;
typedef __attribute__((ext_vector_type(8)))  __bf16 v8bf;

union BF16x16 { v16bf v; v8bf h[2]; __bf16 e[16]; };

// Fragment loader for bf16 data stored K-contiguous (A: [M][K] row-major,
// B: transposed weights [N][K]).  Wave32 WMMA 16-bit operand layout:
//   lanes 0-15 : row/col rc0+lane,    K = k0+{0..7, 16..23}
//   lanes 16-31: row/col rc0+lane-16, K = k0+{8..15, 24..31}
__device__ __forceinline__ v16bf load_frag_bf16(const __bf16* __restrict__ p,
                                                long ld, long rc0, int k0) {
    int  lane = threadIdx.x & 31;
    long rc   = rc0 + (lane & 15);
    int  kb   = k0 + ((lane >> 4) << 3);
    BF16x16 f;
    f.h[0] = *(const v8bf*)(p + rc * ld + kb);
    f.h[1] = *(const v8bf*)(p + rc * ld + kb + 16);
    return f.v;
}

// A fragment from f32 source (x), converted to bf16 in-register
// (lowers to 8x v_cvt_pk_bf16_f32).
__device__ __forceinline__ v16bf load_a_f32(const float* __restrict__ src,
                                            long ld, long m0, int k0) {
    int  lane = threadIdx.x & 31;
    long row  = m0 + (lane & 15);
    int  kb   = k0 + ((lane >> 4) << 3);
    const float4* p0 = (const float4*)(src + row * ld + kb);
    const float4* p1 = (const float4*)(src + row * ld + kb + 16);
    float4 x0 = p0[0], x1 = p0[1];
    float4 y0 = p1[0], y1 = p1[1];
    BF16x16 f;
    f.e[0]  = (__bf16)x0.x; f.e[1]  = (__bf16)x0.y; f.e[2]  = (__bf16)x0.z; f.e[3]  = (__bf16)x0.w;
    f.e[4]  = (__bf16)x1.x; f.e[5]  = (__bf16)x1.y; f.e[6]  = (__bf16)x1.z; f.e[7]  = (__bf16)x1.w;
    f.e[8]  = (__bf16)y0.x; f.e[9]  = (__bf16)y0.y; f.e[10] = (__bf16)y0.z; f.e[11] = (__bf16)y0.w;
    f.e[12] = (__bf16)y1.x; f.e[13] = (__bf16)y1.y; f.e[14] = (__bf16)y1.z; f.e[15] = (__bf16)y1.w;
    return f.v;
}

#define WMMA_BF16(A, B, C) \
    __builtin_amdgcn_wmma_f32_16x16x32_bf16(false, (A), false, (B), (short)0, (C), false, false)

// W [K][N] f32 row-major -> WT [N][K] bf16
__global__ __launch_bounds__(256) void convert_transpose(const float* __restrict__ W,
                                                         __bf16* __restrict__ WT,
                                                         int K, int N) {
    long idx = (long)blockIdx.x * blockDim.x + threadIdx.x;
    if (idx >= (long)K * N) return;
    int k = (int)(idx / N);
    int n = (int)(idx - (long)k * N);
    WT[(long)n * K + k] = (__bf16)W[idx];
}

// xp[bt][h] = bf16( x[bt][:] @ Wxh[:][h] + b[h] )
// Each wave computes a 64x16 output strip (4 M-subtiles share one B fragment).
// K pipelined 2 deep (chunks of 64) with independent register sets to kill
// the WMMA->VALU WAR-hazard NOPs.
__global__ __launch_bounds__(256) void xp_gemm(const float* __restrict__ x,
                                               const __bf16* __restrict__ WxhT,
                                               const float* __restrict__ bias,
                                               __bf16* __restrict__ xp) {
    int  wave = threadIdx.x >> 5;
    long wid  = (long)blockIdx.x * 8 + wave;      // 0 .. 65535
    long m0   = (wid >> 6) << 6;                  // 64-row supertile over M=65536
    int  n0   = (int)(wid & 63) << 4;             // 16-col tile over N=1024

    v8f acc[4];
    acc[0] = {}; acc[1] = {}; acc[2] = {}; acc[3] = {};

    v16bf bfA, bfB, afA[4], afB[4];
    bfA = load_frag_bf16(WxhT, Dsz, n0, 0);
#pragma unroll
    for (int s = 0; s < 4; ++s) afA[s] = load_a_f32(x, Dsz, m0 + s * 16, 0);

    for (int k0 = 0; k0 < Dsz; k0 += 64) {
        // prefetch+convert chunk k0+32 into the B register set
        bfB = load_frag_bf16(WxhT, Dsz, n0, k0 + 32);
#pragma unroll
        for (int s = 0; s < 4; ++s) afB[s] = load_a_f32(x, Dsz, m0 + s * 16, k0 + 32);
#pragma unroll
        for (int s = 0; s < 4; ++s) acc[s] = WMMA_BF16(afA[s], bfA, acc[s]);

        if (k0 + 64 < Dsz) {
            // prefetch+convert chunk k0+64 into the A register set
            bfA = load_frag_bf16(WxhT, Dsz, n0, k0 + 64);
#pragma unroll
            for (int s = 0; s < 4; ++s) afA[s] = load_a_f32(x, Dsz, m0 + s * 16, k0 + 64);
        }
#pragma unroll
        for (int s = 0; s < 4; ++s) acc[s] = WMMA_BF16(afB[s], bfB, acc[s]);
    }

    int   lane = threadIdx.x & 31;
    int   n    = n0 + (lane & 15);
    int   rb   = (lane >> 4) << 3;                // lanes>=16 hold rows M=8..15
    float bv   = bias[n];
#pragma unroll
    for (int s = 0; s < 4; ++s) {
#pragma unroll
        for (int r = 0; r < 8; ++r) {
            long m = m0 + s * 16 + rb + r;
            xp[m * (long)Hsz + n] = (__bf16)(acc[s][r] + bv);
        }
    }
}

// h_next = tanh(xp[:, t, :] + h_prev @ Whh)   (one 16x16 tile per wave)
// h kept in bf16; writes f32 result to `out` on the final step.
__global__ __launch_bounds__(256) void rnn_step(const __bf16* __restrict__ hprev,
                                                const __bf16* __restrict__ WhhT,
                                                const __bf16* __restrict__ xp,
                                                int t,
                                                __bf16* __restrict__ hnext,
                                                float* __restrict__ out) {
    int wave = threadIdx.x >> 5;
    int wid  = blockIdx.x * 8 + wave;             // 0 .. 1023
    int m0   = (wid >> 6) << 4;                   // 16 tiles over M=256
    int n0   = (wid & 63) << 4;                   // 64 tiles over N=1024

    v8f acc = {};
    v16bf aA = load_frag_bf16(hprev, Hsz, m0, 0);
    v16bf bA = load_frag_bf16(WhhT, Hsz, n0, 0);

    for (int k0 = 0; k0 < Hsz; k0 += 64) {
        v16bf aB = load_frag_bf16(hprev, Hsz, m0, k0 + 32);
        v16bf bB = load_frag_bf16(WhhT, Hsz, n0, k0 + 32);
        acc = WMMA_BF16(aA, bA, acc);
        if (k0 + 64 < Hsz) {
            aA = load_frag_bf16(hprev, Hsz, m0, k0 + 64);
            bA = load_frag_bf16(WhhT, Hsz, n0, k0 + 64);
        }
        acc = WMMA_BF16(aB, bB, acc);
    }

    int lane = threadIdx.x & 31;
    int n    = n0 + (lane & 15);
    int rb   = (lane >> 4) << 3;
#pragma unroll
    for (int r = 0; r < 8; ++r) {
        int   b   = m0 + rb + r;                  // batch index
        float pre = acc[r] + (float)xp[((long)b * Tsz + t) * Hsz + n];
        float hv  = tanhf(pre);
        hnext[(long)b * Hsz + n] = (__bf16)hv;
        if (out) out[(long)b * Hsz + n] = hv;
    }
}

extern "C" void kernel_launch(void* const* d_in, const int* in_sizes, int n_in,
                              void* d_out, int out_size, void* d_ws, size_t ws_size,
                              hipStream_t stream) {
    (void)in_sizes; (void)n_in; (void)out_size; (void)ws_size;
    const float* x    = (const float*)d_in[0];   // [B,T,D]
    const float* Wxh  = (const float*)d_in[1];   // [D,H]
    const float* Whh  = (const float*)d_in[2];   // [H,H]
    const float* bias = (const float*)d_in[3];   // [H]
    float* out = (float*)d_out;                  // [B,H]

    char* ws = (char*)d_ws;
    __bf16* WxhT = (__bf16*)(ws);                          // 2 MB bf16 [H][D]
    __bf16* WhhT = (__bf16*)(ws + (2l << 20));             // 2 MB bf16 [H][H]
    __bf16* h0   = (__bf16*)(ws + (4l << 20));             // 512 KB bf16 [B][H]
    __bf16* h1   = (__bf16*)(ws + (4l << 20) + (512l << 10));
    __bf16* xp   = (__bf16*)(ws + (5l << 20));             // 128 MB bf16 [B*T][H]

    hipMemsetAsync(h0, 0, (size_t)Bsz * Hsz * sizeof(__bf16), stream);

    int total = Dsz * Hsz;
    convert_transpose<<<(total + 255) / 256, 256, 0, stream>>>(Wxh, WxhT, Dsz, Hsz);
    convert_transpose<<<(total + 255) / 256, 256, 0, stream>>>(Whh, WhhT, Hsz, Hsz);

    // 65536 wave-jobs / 8 waves per block
    xp_gemm<<<8192, 256, 0, stream>>>(x, WxhT, bias, xp);

    __bf16* bufs[2] = { h0, h1 };
    for (int t = 0; t < Tsz; ++t) {
        const __bf16* hp  = bufs[t & 1];
        __bf16*       hn  = bufs[(t + 1) & 1];
        float*        dst = (t == Tsz - 1) ? out : nullptr;
        rnn_step<<<128, 256, 0, stream>>>(hp, WhhT, xp, t, hn, dst);
    }
}
